// SPINN_77833397338320
// MI455X (gfx1250) — compile-verified, matrix operand order checked by reading, not run on hardware
//
#include <hip/hip_runtime.h>
#include <hip/hip_bf16.h>

// SPINN forward for MI455X (gfx1250, wave32, WMMA).
// B=64, L=32, D=TD=768, T=63, S=32. Out: [64, 768 + 126].

typedef __attribute__((ext_vector_type(16))) __bf16 v16bf;
typedef __attribute__((ext_vector_type(8)))  __bf16 v8bf;
typedef __attribute__((ext_vector_type(8)))  float  v8f;

#define B_   64
#define L_   32
#define D_   768
#define T_   63
#define S_   32
#define N1A  3072          // lstm_in cols
#define N1B  3840          // gate cols
#define N2   3872          // W_track(3840) + W_trans(2) padded to 32-tile
#define K1A  3072          // bh|s1h|s2h|trk_h
#define K1B  1536          // s1h|s2h
#define K2   768
#define OUTLD 894          // D + 2T

__device__ __forceinline__ float sigf(float x) { return 1.0f / (1.0f + __expf(-x)); }
__device__ __forceinline__ int   clampi(int v, int lo, int hi) { return v < lo ? lo : (v > hi ? hi : v); }

// ---------------- zero fill (16B granules) ----------------
__global__ void zero_kernel(uint4* __restrict__ p, long n16) {
  long i = (long)blockIdx.x * blockDim.x + threadIdx.x;
  if (i < n16) { uint4 z; z.x = z.y = z.z = z.w = 0u; p[i] = z; }
}

// ---------------- fp32 -> bf16 transpose (LDS-tiled) ----------------
// dst[n * dst_ld + k_off + k] = (bf16) src[k * src_ld + n]
__global__ __launch_bounds__(256) void transpose_cvt(
    const float* __restrict__ src, int src_ld, int rows /*K*/, int cols /*N*/,
    __bf16* __restrict__ dst, int dst_ld, int dst_k_off) {
  __shared__ float tile[32][33];
  int kb = blockIdx.y * 32, nb = blockIdx.x * 32;
  int tx = threadIdx.x, ty = threadIdx.y;           // 32 x 8
  #pragma unroll
  for (int i = ty; i < 32; i += 8) {
    int k = kb + i, n = nb + tx;
    tile[i][tx] = (k < rows && n < cols) ? src[(long)k * src_ld + n] : 0.f;
  }
  __syncthreads();
  #pragma unroll
  for (int i = ty; i < 32; i += 8) {
    int n = nb + i, k = kb + tx;
    if (n < cols && k < rows)
      dst[(long)n * dst_ld + dst_k_off + k] = (__bf16)tile[tx][i];
  }
}

// ---------------- bf16 WMMA GEMM: C[64,N] = A[64,K] @ Wt^T (+bias) ----------------
// Wt stored transposed: Wt[n * K + k].  Grid.x = N/32; 4 waves/block cover M=64.
// Fragment layout per ISA 7.12.2 (16-bit, wave32): lane holds K groups
// {sel*8..sel*8+7} and {sel*8+16..sel*8+23}, sel = lane>>4.
__global__ __launch_bounds__(128) void gemm_bf16_wmma(
    const __bf16* __restrict__ A, int lda,
    const __bf16* __restrict__ Wt, int K,
    const float* __restrict__ bias,
    float* __restrict__ C, int ldc) {
  const int lane = threadIdx.x & 31;
  const int wave = threadIdx.x >> 5;        // m-tile 0..3
  const int nt0  = blockIdx.x * 2;          // two n-tiles per wave
  const int lr   = lane & 15;
  const int sel  = lane >> 4;
  const __bf16* Ap  = A  + (long)(wave * 16 + lr) * lda + sel * 8;
  const __bf16* Bp0 = Wt + (long)(nt0 * 16 + lr) * K   + sel * 8;
  const __bf16* Bp1 = Bp0 + (long)16 * K;
  v8f acc0 = {};
  v8f acc1 = {};
  for (int k0 = 0; k0 < K; k0 += 32) {
    if ((k0 & 255) == 0) {                  // speculative L2 prefetch ahead
      __builtin_prefetch(Bp0 + k0 + 512, 0, 0);
      __builtin_prefetch(Bp1 + k0 + 512, 0, 0);
    }
    union { v16bf v; v8bf h[2]; } a, b0, b1;
    a.h[0]  = *(const v8bf*)(Ap  + k0);
    a.h[1]  = *(const v8bf*)(Ap  + k0 + 16);
    b0.h[0] = *(const v8bf*)(Bp0 + k0);
    b0.h[1] = *(const v8bf*)(Bp0 + k0 + 16);
    b1.h[0] = *(const v8bf*)(Bp1 + k0);
    b1.h[1] = *(const v8bf*)(Bp1 + k0 + 16);
    acc0 = __builtin_amdgcn_wmma_f32_16x16x32_bf16(false, a.v, false, b0.v,
                                                   (short)0, acc0, false, false);
    acc1 = __builtin_amdgcn_wmma_f32_16x16x32_bf16(false, a.v, false, b1.v,
                                                   (short)0, acc1, false, false);
  }
  // C/D layout: VGPR r -> row = wave*16 + sel*8 + r, col = nt*16 + (lane&15)
  const int row  = wave * 16 + sel * 8;
  const int col0 = nt0 * 16 + lr;
  const float bv0 = bias ? bias[col0]      : 0.f;
  const float bv1 = bias ? bias[col0 + 16] : 0.f;
  #pragma unroll
  for (int r = 0; r < 8; ++r) {
    C[(long)(row + r) * ldc + col0]      = acc0[r] + bv0;
    C[(long)(row + r) * ldc + col0 + 16] = acc1[r] + bv1;
  }
}

// ---------------- build bf16 activation row [bh | s1h | s2h | trk_h] ----------------
__global__ __launch_bounds__(256) void build_a1(
    __bf16* __restrict__ A1, const float* __restrict__ buffer_h,
    const float* __restrict__ stack_h, const float* __restrict__ trk_h,
    const int* __restrict__ ptr_in, const int* __restrict__ buft_in) {
  int idx = blockIdx.x * blockDim.x + threadIdx.x;   // 64*3072
  int b = idx / K1A, j = idx - b * K1A;
  int ptr = ptr_in[b];
  float v;
  if (j < 768) {
    int pos = clampi(L_ - 1 - buft_in[b], 0, L_ - 1);
    v = buffer_h[(b * L_ + pos) * D_ + j];
  } else if (j < 1536) {
    int d = j - 768;
    int p1 = clampi(ptr - 1, 0, S_ - 1);
    v = (ptr > 0) ? stack_h[(b * S_ + p1) * D_ + d] : 0.f;
  } else if (j < 2304) {
    int d = j - 1536;
    int p2 = clampi(ptr - 2, 0, S_ - 1);
    v = (ptr > 1) ? stack_h[(b * S_ + p2) * D_ + d] : 0.f;
  } else {
    v = trk_h[b * D_ + (j - 2304)];
  }
  A1[idx] = (__bf16)v;
}

// ---------------- tracker LSTM cell ----------------
__global__ __launch_bounds__(256) void ew_tracker(
    const float* __restrict__ Y1a, float* __restrict__ trk_c,
    float* __restrict__ trk_h, __bf16* __restrict__ trk_h_bf) {
  int idx = blockIdx.x * blockDim.x + threadIdx.x;   // 64*768
  int b = idx / D_, d = idx - b * D_;
  const float* y = Y1a + (long)b * N1A;
  float a = y[d], i = y[768 + d], f = y[1536 + d], o = y[2304 + d];
  float c = tanhf(a) * sigf(i) + sigf(f) * trk_c[idx];
  trk_c[idx] = c;
  float h = sigf(o) * tanhf(c);
  trk_h[idx] = h;
  trk_h_bf[idx] = (__bf16)h;
}

// ---------------- compose + shift/reduce stack update + logits ----------------
__global__ __launch_bounds__(256) void ew_compose(
    const float* __restrict__ Yg, const float* __restrict__ Y2,
    float* __restrict__ stack_h, float* __restrict__ stack_c,
    const float* __restrict__ buffer_h, const float* __restrict__ buffer_c,
    const int* __restrict__ trans, const float* __restrict__ b_trans,
    const int* __restrict__ ptr_in, int* __restrict__ ptr_out,
    const int* __restrict__ buft_in, int* __restrict__ buft_out,
    float* __restrict__ out, int t) {
  int idx = blockIdx.x * blockDim.x + threadIdx.x;   // 64*768
  int b = idx / D_, d = idx - b * D_;
  int ptr = ptr_in[b];
  int buf_t = buft_in[b];
  int tr = trans[b * T_ + t];
  const float* yg = Yg + (long)b * N1B;
  const float* y2 = Y2 + (long)b * N2;
  float gi  = yg[d]        + y2[d];
  float gfl = yg[768 + d]  + y2[768 + d];
  float gfr = yg[1536 + d] + y2[1536 + d];
  float go  = yg[2304 + d] + y2[2304 + d];
  float gg  = yg[3072 + d] + y2[3072 + d];
  int p1 = clampi(ptr - 1, 0, S_ - 1), p2 = clampi(ptr - 2, 0, S_ - 1);
  float s1c = (ptr > 0) ? stack_c[(b * S_ + p1) * D_ + d] : 0.f;
  float s2c = (ptr > 1) ? stack_c[(b * S_ + p2) * D_ + d] : 0.f;
  float comp_c = sigf(gfl) * s2c + sigf(gfr) * s1c + sigf(gi) * tanhf(gg);
  float comp_h = sigf(go) * tanhf(comp_c);
  bool is_shift = (tr == 0), is_red = (tr == 1), is_skip = (tr == 2);
  int buf_pos = clampi(L_ - 1 - buf_t, 0, L_ - 1);
  int wpos = clampi(is_shift ? ptr : ptr - 2, 0, S_ - 1);
  float wh = is_shift ? buffer_h[(b * L_ + buf_pos) * D_ + d] : comp_h;
  float wc = is_shift ? buffer_c[(b * L_ + buf_pos) * D_ + d] : comp_c;
  if (is_skip) {
    wh = stack_h[(b * S_ + wpos) * D_ + d];
    wc = stack_c[(b * S_ + wpos) * D_ + d];
  }
  stack_h[(b * S_ + wpos) * D_ + d] = wh;
  stack_c[(b * S_ + wpos) * D_ + d] = wc;
  if (d == 0) {
    ptr_out[b]  = ptr + (is_shift ? 1 : (is_red ? -1 : 0));
    buft_out[b] = buf_t + (is_shift ? 1 : 0);
  }
  if (d < 2) out[b * OUTLD + D_ + t * 2 + d] = y2[3840 + d] + b_trans[d];
}

// ---------------- final_h gather ----------------
__global__ __launch_bounds__(256) void final_gather(
    float* __restrict__ out, const float* __restrict__ stack_h,
    const int* __restrict__ ptr_in) {
  int idx = blockIdx.x * blockDim.x + threadIdx.x;   // 64*768
  int b = idx / D_, d = idx - b * D_;
  int p = clampi(ptr_in[b] - 1, 0, S_ - 1);
  out[b * OUTLD + d] = stack_h[(b * S_ + p) * D_ + d];
}

extern "C" void kernel_launch(void* const* d_in, const int* in_sizes, int n_in,
                              void* d_out, int out_size, void* d_ws, size_t ws_size,
                              hipStream_t stream) {
  const float* buffer_h = (const float*)d_in[0];
  const float* buffer_c = (const float*)d_in[1];
  const int*   trans    = (const int*)d_in[2];
  const float* W_buf    = (const float*)d_in[3];
  const float* W_s1     = (const float*)d_in[4];
  const float* W_s2     = (const float*)d_in[5];
  const float* W_lat    = (const float*)d_in[6];
  const float* b_lat    = (const float*)d_in[7];
  const float* W_trans  = (const float*)d_in[8];
  const float* b_trans  = (const float*)d_in[9];
  const float* W_left   = (const float*)d_in[10];
  const float* b_left   = (const float*)d_in[11];
  const float* W_right  = (const float*)d_in[12];
  const float* W_track  = (const float*)d_in[13];
  float* out = (float*)d_out;

  // workspace carve-up (256B aligned slices)
  char* ws = (char*)d_ws;
  size_t off = 0;
  auto carve = [&](size_t bytes) -> void* {
    void* p = ws + off;
    off += (bytes + 255) & ~(size_t)255;
    return p;
  };
  float* stack_h  = (float*)carve((size_t)B_ * S_ * D_ * 4);
  float* stack_c  = (float*)carve((size_t)B_ * S_ * D_ * 4);
  float* trk_c    = (float*)carve((size_t)B_ * D_ * 4);
  float* trk_h    = (float*)carve((size_t)B_ * D_ * 4);
  int*   ptrbuf   = (int*)carve(2 * B_ * 4);          // double-buffered
  int*   buftbuf  = (int*)carve(2 * B_ * 4);
  size_t state_bytes = off;                           // zero all of the above
  __bf16* A1      = (__bf16*)carve((size_t)B_ * K1A * 2);
  __bf16* trk_hbf = (__bf16*)carve((size_t)B_ * D_ * 2);
  float* Y1a      = (float*)carve((size_t)B_ * N1A * 4);
  float* Y1b      = (float*)carve((size_t)B_ * N1B * 4);
  float* Y2       = (float*)carve((size_t)B_ * N2  * 4);
  __bf16* Wta     = (__bf16*)carve((size_t)N1A * K1A * 2);
  __bf16* Wtb     = (__bf16*)carve((size_t)N1B * K1B * 2);
  __bf16* Wt2     = (__bf16*)carve((size_t)N2  * K2  * 2);
  (void)in_sizes; (void)n_in; (void)out_size; (void)ws_size;

  // ---- init state (fresh every call: deterministic) ----
  {
    long n16 = (long)(state_bytes / 16);
    zero_kernel<<<(int)((n16 + 255) / 256), 256, 0, stream>>>((uint4*)stack_h, n16);
    long w16 = (long)((size_t)N2 * K2 * 2 / 16);      // zero Wt2 (pad cols -> 0)
    zero_kernel<<<(int)((w16 + 255) / 256), 256, 0, stream>>>((uint4*)Wt2, w16);
  }

  // ---- one-time weight transpose + bf16 convert ----
  dim3 tb(32, 8);
  dim3 g3072((3072 + 31) / 32, (768 + 31) / 32);
  dim3 g3840((3840 + 31) / 32, (768 + 31) / 32);
  dim3 g2(1, (768 + 31) / 32);
  transpose_cvt<<<g3072, tb, 0, stream>>>(W_buf, 3072, 768, 3072, Wta, K1A, 0);
  transpose_cvt<<<g3072, tb, 0, stream>>>(W_s1,  3072, 768, 3072, Wta, K1A, 768);
  transpose_cvt<<<g3072, tb, 0, stream>>>(W_s2,  3072, 768, 3072, Wta, K1A, 1536);
  transpose_cvt<<<g3072, tb, 0, stream>>>(W_lat, 3072, 768, 3072, Wta, K1A, 2304);
  transpose_cvt<<<g3840, tb, 0, stream>>>(W_right, 3840, 768, 3840, Wtb, K1B, 0);
  transpose_cvt<<<g3840, tb, 0, stream>>>(W_left,  3840, 768, 3840, Wtb, K1B, 768);
  transpose_cvt<<<g3840, tb, 0, stream>>>(W_track, 3840, 768, 3840, Wt2, K2, 0);
  transpose_cvt<<<g2,    tb, 0, stream>>>(W_trans, 2,    768, 2, Wt2 + (size_t)3840 * K2, K2, 0);

  // ---- 63 sequential shift-reduce steps ----
  for (int t = 0; t < T_; ++t) {
    int pin = t & 1, pout = pin ^ 1;
    build_a1<<<(B_ * K1A) / 256, 256, 0, stream>>>(
        A1, buffer_h, stack_h, trk_h, ptrbuf + pin * B_, buftbuf + pin * B_);
    // lstm_in = [bh|s1h|s2h|trk_h] @ Wta + b_lat
    gemm_bf16_wmma<<<N1A / 32, 128, 0, stream>>>(A1, K1A, Wta, K1A, b_lat, Y1a, N1A);
    // gates_part = [s1h|s2h] @ Wtb + b_left
    gemm_bf16_wmma<<<N1B / 32, 128, 0, stream>>>(A1 + 768, K1A, Wtb, K1B, b_left, Y1b, N1B);
    ew_tracker<<<(B_ * D_) / 256, 256, 0, stream>>>(Y1a, trk_c, trk_h, trk_hbf);
    // [trk_h @ W_track | trk_h @ W_trans | pad]
    gemm_bf16_wmma<<<N2 / 32, 128, 0, stream>>>(trk_hbf, K2, Wt2, K2, nullptr, Y2, N2);
    ew_compose<<<(B_ * D_) / 256, 256, 0, stream>>>(
        Y1b, Y2, stack_h, stack_c, buffer_h, buffer_c, trans, b_trans,
        ptrbuf + pin * B_, ptrbuf + pout * B_,
        buftbuf + pin * B_, buftbuf + pout * B_, out, t);
  }
  final_gather<<<(B_ * D_) / 256, 256, 0, stream>>>(out, stack_h, ptrbuf + (T_ & 1) * B_);
}
